// BaseTransformer_43078521979301
// MI455X (gfx1250) — matrix operand order repaired and checked by reading, not compiled, not run on hardware
//
#include <hip/hip_runtime.h>

// ---------------------------------------------------------------------------
// CDNA5 (gfx1250, wave32) transformer block:
//   qkv = Wqkv @ x + b ; flash attention (online softmax) ; out = Wo@z + Wres@x + b
// GEMMs: v_wmma_f32_16x16x32_bf16 (f32 accumulate).
// Attention K/V staged into LDS once per workgroup via the Tensor Data Mover
// (tensor_load_to_lds + s_wait_tensorcnt), then consumed with ds_load_b128.
// ---------------------------------------------------------------------------

typedef __attribute__((ext_vector_type(16))) __bf16 bf16x16;
typedef __attribute__((ext_vector_type(8)))  float  f32x8;
typedef __attribute__((ext_vector_type(4)))  unsigned int u32x4;
typedef __attribute__((ext_vector_type(8)))  int  i32x8;
typedef __attribute__((ext_vector_type(4)))  int  i32x4;

#define WMMA_BF16(A, B, C) \
  __builtin_amdgcn_wmma_f32_16x16x32_bf16(false, (A), false, (B), (short)0, (C), false, false)

namespace {
constexpr int kB    = 16;
constexpr int kCin  = 128;
constexpr int kCout = 128;
constexpr int kL    = 1024;
constexpr int kDK   = 32;
constexpr int kDV   = 32;
constexpr int kH    = 8;
constexpr int kHD   = kH * kDV;                 // 256 (z channel dim)
constexpr float kScale = 0.17677669529663688f;  // DK^-0.5

union Op16 { uint4 u4[2]; bf16x16 v; };
} // namespace

// TDM: flat copy of `bytes` (multiple of 8, <= 512KB) from global to LDS.
// D# per ISA 8.3/8.4: 1 row of n8 8-byte elements; groups 2/3 zero (<=2D).
// This toolchain exposes the 6-arg builtin: (g0, g1, g2, g3, g4, cpol).
__device__ __forceinline__ void tdm_copy_to_lds(const void* gsrc, unsigned lds_off,
                                                unsigned bytes) {
  unsigned n8 = bytes >> 3;                          // 8-byte elements
  unsigned long long ga = (unsigned long long)gsrc;
  u32x4 g0 = (u32x4){
      1u,                                            // count=1, user desc
      lds_off,                                       // lds_addr (bits 63:32)
      (unsigned)ga,                                  // global_addr[31:0]
      (((unsigned)(ga >> 32)) & 0x01FFFFFFu) | 0x80000000u  // addr[56:32] | type=2
  };
  i32x8 g1 = (i32x8){
      0x00030000,                                    // data_size=8B; no mask/pad/iter
      (int)((n8 & 0xFFFFu) << 16),                   // tensor_dim0[15:0]
      (int)((n8 >> 16) | (1u << 16)),                // tensor_dim0[31:16] | tensor_dim1=1
      (int)((n8 & 0xFFFFu) << 16),                   // tile_dim0 = n8
      1,                                             // tile_dim1 = 1
      (int)n8,                                       // tensor_dim0_stride[31:0]
      0, 0
  };
  i32x4 z4 = (i32x4){0, 0, 0, 0};
  i32x8 z8 = (i32x8){0, 0, 0, 0, 0, 0, 0, 0};
  __builtin_amdgcn_tensor_load_to_lds(g0, g1, z4, z4, z8, 0);
}

// ---- prep: x[b][c][l] (f32) -> xt[b][l][c] (bf16) --------------------------
__global__ void k_prep_x(const float* __restrict__ x, __bf16* __restrict__ xt) {
  int idx = blockIdx.x * blockDim.x + threadIdx.x;   // over B*Cin*L, coalesced read
  int l = idx % kL;
  int c = (idx / kL) % kCin;
  int b = idx / (kL * kCin);
  xt[((size_t)b * kL + l) * kCin + c] = (__bf16)x[idx];
}

// ---- prep: weight f32 -> bf16 (row-major kept) -----------------------------
__global__ void k_prep_w(const float* __restrict__ wq, const float* __restrict__ wo,
                         const float* __restrict__ wr,
                         __bf16* __restrict__ wq_b, __bf16* __restrict__ wo_b,
                         __bf16* __restrict__ wr_b) {
  int idx = blockIdx.x * blockDim.x + threadIdx.x;
  const int n_q = 768 * 128, n_o = 128 * 256, n_r = 128 * 128;
  if (idx < n_q)                  wq_b[idx]              = (__bf16)wq[idx];
  else if (idx < n_q + n_o)       wo_b[idx - n_q]        = (__bf16)wo[idx - n_q];
  else if (idx < n_q + n_o + n_r) wr_b[idx - n_q - n_o]  = (__bf16)wr[idx - n_q - n_o];
}

// ---- QKV projection: one wave = one 16(o) x 16(l) tile, K=128 --------------
// o-tile j within a head: 0,1 -> Q; 2,3 -> K; 4,5 -> V.
// Q,K stored [b][h][s][d] (d contiguous); V stored [b][h][d][t] (t contiguous).
__global__ void k_qkv(const __bf16* __restrict__ w, const float* __restrict__ bias,
                      const __bf16* __restrict__ xt,
                      __bf16* __restrict__ q, __bf16* __restrict__ k,
                      __bf16* __restrict__ v) {
  int wave = __builtin_amdgcn_readfirstlane(
      (int)((blockIdx.x * blockDim.x + threadIdx.x) >> 5));
  int lane = threadIdx.x & 31;
  int lt = wave & 63;                   // 64 l-tiles
  int j  = (wave >> 6) % 6;             // 6 o-tiles per head
  int rest = wave / (64 * 6);
  int h = rest & 7;
  int b = rest >> 3;
  int o0 = h * 96 + j * 16;
  int l0 = lt * 16;
  int m = lane & 15, half = lane >> 4;

  const __bf16* arow = w  + (size_t)(o0 + m) * kCin;            // A row o0+m
  const __bf16* brow = xt + ((size_t)b * kL + l0 + m) * kCin;   // B col l0+m
  f32x8 acc = {};
#pragma unroll
  for (int kc = 0; kc < kCin; kc += 32) {
    Op16 a, bb;
    a.u4[0]  = *(const uint4*)(arow + kc + half * 16);
    a.u4[1]  = *(const uint4*)(arow + kc + half * 16 + 8);
    bb.u4[0] = *(const uint4*)(brow + kc + half * 16);
    bb.u4[1] = *(const uint4*)(brow + kc + half * 16 + 8);
    acc = WMMA_BF16(a.v, bb.v, acc);
  }
#pragma unroll
  for (int r = 0; r < 8; ++r) {
    int mr = r + 8 * half;              // row within tile (C layout)
    int o  = o0 + mr;
    int s  = l0 + m;                    // col = seq position
    float val = acc[r] + bias[o];
    if (j < 2) {
      int d = j * 16 + mr;
      q[(((size_t)b * kH + h) * kL + s) * kDK + d] = (__bf16)val;
    } else if (j < 4) {
      int d = (j - 2) * 16 + mr;
      k[(((size_t)b * kH + h) * kL + s) * kDK + d] = (__bf16)val;
    } else {
      int d = (j - 4) * 16 + mr;
      v[(((size_t)b * kH + h) * kDV + d) * kL + s] = (__bf16)val;
    }
  }
}

// ---- Attention: one block = 8 waves = 8 s-tiles of one (b,h,s-group) -------
// K (64KB) + V (64KB) for the head are DMA'd to LDS once by wave 0 (TDM),
// then every WMMA B-operand is a ds_load_b128.
__global__ void k_attn(const __bf16* __restrict__ q, const __bf16* __restrict__ kk,
                       const __bf16* __restrict__ v, __bf16* __restrict__ zt) {
  extern __shared__ __align__(16) char dsm[];
  __bf16* kl = (__bf16*)dsm;                       // [t][d]  64 KB
  __bf16* vl = (__bf16*)(dsm + 65536);             // [d][t]  64 KB
  __bf16* pb = (__bf16*)(dsm + 131072);            // P tiles, 1 KB/wave

  int wv   = __builtin_amdgcn_readfirstlane((int)(threadIdx.x >> 5));
  int lane = threadIdx.x & 31;
  int sg = blockIdx.x & 7;                         // s-group (8 tiles each)
  int h  = (blockIdx.x >> 3) & 7;
  int b  = blockIdx.x >> 6;
  int s0 = sg * 128 + wv * 16;
  int m = lane & 15, half = lane >> 4;

  const __bf16* qbh = q  + (size_t)(b * kH + h) * kL * kDK;
  const __bf16* kbh = kk + (size_t)(b * kH + h) * kL * kDK;
  const __bf16* vbh = v  + (size_t)(b * kH + h) * kDV * kL;
  __bf16* pl = pb + wv * (16 * 32);

  // Stage K and V for this head into LDS (one DMA pair per workgroup).
  if (threadIdx.x < 32) {
    unsigned base = (unsigned)(unsigned long long)(void*)dsm;  // LDS byte offset
    tdm_copy_to_lds(kbh, base,          kL * kDK * 2);
    tdm_copy_to_lds(vbh, base + 65536,  kDV * kL * 2);
    __builtin_amdgcn_s_wait_tensorcnt(0);
  }
  __syncthreads();

  // Q A-operand (16 s-rows x 32 d), held in VGPRs for the whole loop.
  Op16 aq;
  aq.u4[0] = *(const uint4*)(qbh + (size_t)(s0 + m) * kDK + half * 16);
  aq.u4[1] = *(const uint4*)(qbh + (size_t)(s0 + m) * kDK + half * 16 + 8);

  f32x8 acc0 = {}, acc1 = {};                      // z accumulators d 0-15 / 16-31
  float mi[8], li[8];
#pragma unroll
  for (int r = 0; r < 8; ++r) { mi[r] = -1e30f; li[r] = 0.f; }

  for (int t0 = 0; t0 < kL; t0 += 32) {
    Op16 bk0, bk1;
    bk0.u4[0] = *(const uint4*)(kl + (t0 + m) * kDK + half * 16);
    bk0.u4[1] = *(const uint4*)(kl + (t0 + m) * kDK + half * 16 + 8);
    bk1.u4[0] = *(const uint4*)(kl + (t0 + 16 + m) * kDK + half * 16);
    bk1.u4[1] = *(const uint4*)(kl + (t0 + 16 + m) * kDK + half * 16 + 8);
    f32x8 zz = {};
    f32x8 sA = WMMA_BF16(aq.v, bk0.v, zz);
    f32x8 sB = WMMA_BF16(aq.v, bk1.v, zz);

    // Online softmax; row m=r+8*half lives in one 16-lane half (C layout).
#pragma unroll
    for (int r = 0; r < 8; ++r) {
      float e0 = sA[r] * kScale, e1 = sB[r] * kScale;
      float mx = fmaxf(e0, e1);
#pragma unroll
      for (int off = 1; off < 16; off <<= 1) mx = fmaxf(mx, __shfl_xor(mx, off, 32));
      float mn = fmaxf(mi[r], mx);
      float sc = __expf(mi[r] - mn);
      float p0 = __expf(e0 - mn);
      float p1 = __expf(e1 - mn);
      float rs = p0 + p1;
#pragma unroll
      for (int off = 1; off < 16; off <<= 1) rs += __shfl_xor(rs, off, 32);
      li[r] = li[r] * sc + rs;
      mi[r] = mn;
      acc0[r] *= sc;
      acc1[r] *= sc;
      int mr = r + 8 * half;
      pl[mr * 32 + m]      = (__bf16)p0;           // P in [s][t32] for A reload
      pl[mr * 32 + 16 + m] = (__bf16)p1;
    }
    asm volatile("" ::: "memory");                 // same-wave LDS ops are in-order
    Op16 pa;
    pa.u4[0] = *(const uint4*)(pl + m * 32 + half * 16);
    pa.u4[1] = *(const uint4*)(pl + m * 32 + half * 16 + 8);
    asm volatile("" ::: "memory");

    Op16 bv0, bv1;
    bv0.u4[0] = *(const uint4*)(vl + m * kL + t0 + half * 16);
    bv0.u4[1] = *(const uint4*)(vl + m * kL + t0 + half * 16 + 8);
    bv1.u4[0] = *(const uint4*)(vl + (16 + m) * kL + t0 + half * 16);
    bv1.u4[1] = *(const uint4*)(vl + (16 + m) * kL + t0 + half * 16 + 8);
    acc0 = WMMA_BF16(pa.v, bv0.v, acc0);
    acc1 = WMMA_BF16(pa.v, bv1.v, acc1);
  }

  // z_t[b][s][h*32 + d] (bf16, c contiguous) for the output GEMM B-operand.
#pragma unroll
  for (int r = 0; r < 8; ++r) {
    float inv = 1.0f / li[r];
    int mr = r + 8 * half;
    size_t base = ((size_t)b * kL + (s0 + mr)) * kHD + h * kDV;
    zt[base + m]      = (__bf16)(acc0[r] * inv);
    zt[base + 16 + m] = (__bf16)(acc1[r] * inv);
  }
}

// ---- Output: out = Wo @ z + Wres @ x + (bo+br); one wave = 16x16 tile ------
__global__ void k_out(const __bf16* __restrict__ wo, const __bf16* __restrict__ wr,
                      const float* __restrict__ bo, const float* __restrict__ br,
                      const __bf16* __restrict__ zt, const __bf16* __restrict__ xt,
                      float* __restrict__ out) {
  int wave = __builtin_amdgcn_readfirstlane(
      (int)((blockIdx.x * blockDim.x + threadIdx.x) >> 5));
  int lane = threadIdx.x & 31;
  int lt = wave & 63;
  int ot = (wave >> 6) & 7;
  int b  = wave >> 9;
  int o0 = ot * 16, l0 = lt * 16;
  int m = lane & 15, half = lane >> 4;

  f32x8 acc = {};
  const __bf16* ar  = wr + (size_t)(o0 + m) * kCin;             // residual A
  const __bf16* bxr = xt + ((size_t)b * kL + l0 + m) * kCin;    // residual B
#pragma unroll
  for (int kc = 0; kc < kCin; kc += 32) {
    Op16 a, bb;
    a.u4[0]  = *(const uint4*)(ar + kc + half * 16);
    a.u4[1]  = *(const uint4*)(ar + kc + half * 16 + 8);
    bb.u4[0] = *(const uint4*)(bxr + kc + half * 16);
    bb.u4[1] = *(const uint4*)(bxr + kc + half * 16 + 8);
    acc = WMMA_BF16(a.v, bb.v, acc);
  }
  const __bf16* ao  = wo + (size_t)(o0 + m) * kHD;              // main A
  const __bf16* bzr = zt + ((size_t)b * kL + l0 + m) * kHD;     // main B
#pragma unroll
  for (int kc = 0; kc < kHD; kc += 32) {
    Op16 a, bb;
    a.u4[0]  = *(const uint4*)(ao + kc + half * 16);
    a.u4[1]  = *(const uint4*)(ao + kc + half * 16 + 8);
    bb.u4[0] = *(const uint4*)(bzr + kc + half * 16);
    bb.u4[1] = *(const uint4*)(bzr + kc + half * 16 + 8);
    acc = WMMA_BF16(a.v, bb.v, acc);
  }
#pragma unroll
  for (int r = 0; r < 8; ++r) {
    int mr = r + 8 * half;
    int o  = o0 + mr;
    out[((size_t)b * kCout + o) * kL + (l0 + m)] = acc[r] + bo[o] + br[o];
  }
}

// ---------------------------------------------------------------------------
extern "C" void kernel_launch(void* const* d_in, const int* in_sizes, int n_in,
                              void* d_out, int out_size, void* d_ws, size_t ws_size,
                              hipStream_t stream) {
  const float* x     = (const float*)d_in[0];
  const float* w_qkv = (const float*)d_in[1];
  const float* b_qkv = (const float*)d_in[2];
  const float* w_o   = (const float*)d_in[3];
  const float* b_o   = (const float*)d_in[4];
  const float* w_res = (const float*)d_in[5];
  const float* b_res = (const float*)d_in[6];
  float* out = (float*)d_out;

  // Workspace carve-up (bf16 buffers, all offsets 256B-aligned).
  char* ws = (char*)d_ws;
  size_t off = 0;
  __bf16* xt   = (__bf16*)(ws + off); off += (size_t)kB * kL * kCin * 2;      // 4 MB
  __bf16* wq_b = (__bf16*)(ws + off); off += (size_t)768 * 128 * 2;
  __bf16* wo_b = (__bf16*)(ws + off); off += (size_t)128 * 256 * 2;
  __bf16* wr_b = (__bf16*)(ws + off); off += (size_t)128 * 128 * 2;
  __bf16* qb   = (__bf16*)(ws + off); off += (size_t)kB * kH * kL * kDK * 2;  // 8 MB
  __bf16* kb   = (__bf16*)(ws + off); off += (size_t)kB * kH * kL * kDK * 2;  // 8 MB
  __bf16* vb   = (__bf16*)(ws + off); off += (size_t)kB * kH * kDV * kL * 2;  // 8 MB
  __bf16* zt   = (__bf16*)(ws + off); off += (size_t)kB * kL * kHD * 2;       // 8 MB
  (void)ws_size; (void)in_sizes; (void)n_in; (void)out_size;

  // 1) transpose+convert x, convert weights
  k_prep_x<<<(kB * kCin * kL) / 256, 256, 0, stream>>>(x, xt);
  k_prep_w<<<(768 * 128 + 128 * 256 + 128 * 128) / 256, 256, 0, stream>>>(
      w_qkv, w_o, w_res, wq_b, wo_b, wr_b);

  // 2) QKV projection: 16b * 8h * 6 o-tiles * 64 l-tiles waves, 8 waves/block
  k_qkv<<<(kB * kH * 6 * 64) / 8, 256, 0, stream>>>(wq_b, b_qkv, xt, qb, kb, vb);

  // 3) attention: 1024 blocks; 136 KB dynamic LDS (K 64K + V 64K + P 8K)
  k_attn<<<kB * kH * 8, 256, 136 * 1024, stream>>>(qb, kb, vb, zt);

  // 4) output + residual projection: 16b * 8 o-tiles * 64 l-tiles waves
  k_out<<<(kB * 8 * 64) / 8, 256, 0, stream>>>(wo_b, wr_b, b_o, b_res, zt, xt, out);
}
